// MultiHeadSlotAttentionRef_4355096838597
// MI455X (gfx1250) — compile-verified, matrix operand order checked by reading, not run on hardware
//
#include <hip/hip_runtime.h>
#include <hip/hip_bf16.h>
#include <math.h>

typedef __bf16 bf16_t;
typedef bf16_t v16bf __attribute__((ext_vector_type(16)));
typedef bf16_t v8bf  __attribute__((ext_vector_type(8)));
typedef float  v8f   __attribute__((ext_vector_type(8)));
typedef float  v4f   __attribute__((ext_vector_type(4)));
typedef unsigned v4u __attribute__((ext_vector_type(4)));
typedef unsigned v8u __attribute__((ext_vector_type(8)));

#define B_     32
#define N_     4096
#define D_     256
#define HEADS_ 4
#define DH_    64
#define DI_    256
#define SLOTS_ 8
#define ITERS_ 3
#define HID_   128
#define EPS_   1e-8f

__device__ inline float warp_sum(float v) {
#pragma unroll
  for (int m = 16; m > 0; m >>= 1) v += __shfl_xor(v, m, 32);
  return v;
}

__device__ inline v16bf frag16(v8bf lo, v8bf hi) {
  return __builtin_shufflevector(lo, hi, 0,1,2,3,4,5,6,7,8,9,10,11,12,13,14,15);
}

__device__ inline v8f wmma_bf16(v16bf a, v16bf b, v8f c) {
  return __builtin_amdgcn_wmma_f32_16x16x32_bf16(false, a, false, b, (short)0, c, false, false);
}

// ---------------------------------------------------------------------------
// Weight transpose + bf16 convert: WkT/WvT[di][k] = W[k][di]
// ---------------------------------------------------------------------------
__global__ void convert_weights(const float* __restrict__ Wk, const float* __restrict__ Wv,
                                bf16_t* __restrict__ WkT, bf16_t* __restrict__ WvT) {
  int idx = blockIdx.x * blockDim.x + threadIdx.x;
  if (idx >= D_ * DI_) return;
  int di = idx >> 8, k = idx & 255;
  WkT[di * D_ + k] = (bf16_t)Wk[k * DI_ + di];
  WvT[di * D_ + k] = (bf16_t)Wv[k * DI_ + di];
}

// ---------------------------------------------------------------------------
// Slot init: slots = mu + exp(logsigma) * gaussian(hash)
// ---------------------------------------------------------------------------
__device__ inline unsigned wang(unsigned s) {
  s = (s ^ 61u) ^ (s >> 16); s *= 9u; s ^= s >> 4; s *= 0x27d4eb2du; s ^= s >> 15;
  return s;
}
__global__ void init_slots(const float* __restrict__ mu, const float* __restrict__ logsig,
                           float* __restrict__ slots) {
  int idx = blockIdx.x * blockDim.x + threadIdx.x;
  if (idx >= B_ * SLOTS_ * D_) return;
  int d = idx & 255;
  unsigned h1 = wang((unsigned)idx * 2654435761u + 12345u);
  unsigned h2 = wang(h1 ^ 0x9e3779b9u);
  float u1 = (h1 >> 8) * (1.0f / 16777216.0f) + 1e-7f;
  float u2 = (h2 >> 8) * (1.0f / 16777216.0f);
  float z = sqrtf(-2.0f * __logf(u1)) * __cosf(6.2831853f * u2);
  slots[idx] = mu[d] + __expf(logsig[d]) * z;
}

// ---------------------------------------------------------------------------
// Per-iteration scratch zeroing (updates accumulator, rowsums, padded q)
// ---------------------------------------------------------------------------
__global__ void zero_ws(float* __restrict__ upd, float* __restrict__ rowsum,
                        unsigned* __restrict__ qwords) {
  int idx = blockIdx.x * blockDim.x + threadIdx.x;
  if (idx < B_ * SLOTS_ * DI_) upd[idx] = 0.0f;
  if (idx < B_ * HEADS_ * SLOTS_) rowsum[idx] = 0.0f;
  if (idx < (B_ * HEADS_ * 16 * DH_) / 2) qwords[idx] = 0u;
}

// ---------------------------------------------------------------------------
// Fused LayerNorm + K/V projection (bf16 WMMA) with TDM weight staging.
// Block: 32 tokens, 8 waves = (matrix K/V) x (M-tile 0/1) x (N-half 0/1).
// Per k-step the 256x32 bf16 chunks of WkT and WvT are DMA'd to LDS once
// (tensor_load_to_lds, TENSORcnt), with TDM padding (4 DW per 16 DW) giving
// an 80B LDS row stride -> conflict-free ds_load_b128 B-fragments.
// K stored [b*N+n][di] bf16 ; V stored transposed [b][h][dh][n] bf16.
// ---------------------------------------------------------------------------
#define XLN_STRIDE 264   // 256 + 8 pad elements
#define WST_STRIDE 40    // 32 + 8 pad elements (matches TDM pad: 16DW + 4DW)
__global__ __launch_bounds__(256)
void ln_kv_proj(const float* __restrict__ x, const float* __restrict__ lw,
                const float* __restrict__ lb,
                const bf16_t* __restrict__ WkT, const bf16_t* __restrict__ WvT,
                bf16_t* __restrict__ Kbf, bf16_t* __restrict__ Vt) {
  __shared__ bf16_t xln[32 * XLN_STRIDE];          // 16.9 KB
  __shared__ bf16_t wstage[2][256 * WST_STRIDE];   // 2 x 20 KB (Wk, Wv chunk)
  const int tid = threadIdx.x, wave = tid >> 5, lane = tid & 31;
  const long R0 = (long)blockIdx.x * 32;

  // ---- LayerNorm: each wave normalizes 4 rows, 8 elems/lane ----
  for (int rr = 0; rr < 4; ++rr) {
    int lrow = wave * 4 + rr;
    const float* xp = x + (R0 + lrow) * D_ + lane * 8;
    v4f a = *(const v4f*)xp;
    v4f b = *(const v4f*)(xp + 4);
    float s = 0.f, ss = 0.f;
#pragma unroll
    for (int j = 0; j < 4; ++j) { s += a[j] + b[j]; ss += a[j]*a[j] + b[j]*b[j]; }
    s = warp_sum(s); ss = warp_sum(ss);
    float mean = s * (1.0f / D_);
    float var  = ss * (1.0f / D_) - mean * mean;
    float inv  = rsqrtf(var + 1e-5f);
    v4f w0 = *(const v4f*)(lw + lane * 8);
    v4f w1 = *(const v4f*)(lw + lane * 8 + 4);
    v4f c0 = *(const v4f*)(lb + lane * 8);
    v4f c1 = *(const v4f*)(lb + lane * 8 + 4);
    v8bf o;
#pragma unroll
    for (int j = 0; j < 4; ++j) {
      o[j]     = (bf16_t)((a[j] - mean) * inv * w0[j] + c0[j]);
      o[j + 4] = (bf16_t)((b[j] - mean) * inv * w1[j] + c1[j]);
    }
    *(v8bf*)&xln[lrow * XLN_STRIDE + lane * 8] = o;
  }

  // ---- wave mapping ----
  const int mat    = wave >> 2;        // 0 = K, 1 = V
  const int mt     = (wave >> 1) & 1;  // M-tile within 32 rows
  const int ntbase = (wave & 1) * 8;   // N-half (8 tiles of 16)
  const int lmod = lane & 15;
  const int koff = (lane < 16) ? 0 : 8;
  v8f acc[8] = {};
  const bf16_t* arow = &xln[(mt * 16 + lmod) * XLN_STRIDE];

  // TDM D# group1 (constant): data_size=2B, pad_enable, interval=16DW, amount=4DW,
  // tensor_dim0=32, tensor_dim1=256, tile=32x256, dim0_stride=256 elements.
  v8u g1;
  g1[0] = (1u << 16) | (1u << 20) | (3u << 22) | (3u << 25);
  g1[1] = (32u << 16);
  g1[2] = (256u << 16);
  g1[3] = (32u << 16);
  g1[4] = 256u;
  g1[5] = 256u;
  g1[6] = 0u;
  g1[7] = 0u;
  const unsigned lds0 = (unsigned)(unsigned long long)(size_t)&wstage[0][0];
  const unsigned lds1 = (unsigned)(unsigned long long)(size_t)&wstage[1][0];
  const unsigned long long ga_k = (unsigned long long)(size_t)WkT;
  const unsigned long long ga_v = (unsigned long long)(size_t)WvT;

  for (int kc = 0; kc < 256; kc += 32) {
    __syncthreads();  // xln ready / previous chunk fully consumed
    if (wave == 0) {
      unsigned long long gk = ga_k + (unsigned)(kc * 2);
      unsigned long long gv = ga_v + (unsigned)(kc * 2);
      v4u g0k, g0v;
      g0k[0] = 1u; g0k[1] = lds0;
      g0k[2] = (unsigned)gk;
      g0k[3] = ((unsigned)(gk >> 32) & 0x01FFFFFFu) | 0x80000000u;  // type=2
      g0v[0] = 1u; g0v[1] = lds1;
      g0v[2] = (unsigned)gv;
      g0v[3] = ((unsigned)(gv >> 32) & 0x01FFFFFFu) | 0x80000000u;
      asm volatile("tensor_load_to_lds %0, %1" :: "s"(g0k), "s"(g1) : "memory");
      asm volatile("tensor_load_to_lds %0, %1" :: "s"(g0v), "s"(g1) : "memory");
      __builtin_amdgcn_s_wait_tensorcnt(0);
    }
    __syncthreads();  // staged weights visible to all waves

    v16bf A = frag16(*(const v8bf*)&arow[kc + koff], *(const v8bf*)&arow[kc + koff + 16]);
#pragma unroll
    for (int nt = 0; nt < 8; ++nt) {
      const bf16_t* bp = &wstage[mat][((ntbase + nt) * 16 + lmod) * WST_STRIDE];
      v16bf Bf = frag16(*(const v8bf*)&bp[koff], *(const v8bf*)&bp[koff + 16]);
      acc[nt] = wmma_bf16(A, Bf, acc[nt]);
    }
  }

  // ---- store (C layout: VGPR i -> M=i / i+8) ----
#pragma unroll
  for (int nt = 0; nt < 8; ++nt) {
#pragma unroll
    for (int vi = 0; vi < 8; ++vi) {
      long g = R0 + mt * 16 + vi + ((lane < 16) ? 0 : 8);
      int col = (ntbase + nt) * 16 + lmod;
      bf16_t val = (bf16_t)acc[nt][vi];
      if (mat == 0) {
        Kbf[g * DI_ + col] = val;
      } else {
        long b = g >> 12, n = g & 4095;
        int  h = col >> 6, dh = col & 63;
        Vt[((b * HEADS_ + h) * DH_ + dh) * (long)N_ + n] = val;
      }
    }
  }
}

// ---------------------------------------------------------------------------
// Slot LN + q projection. q stored bf16 [b][h][16 slots (8..15 zero)][dh].
// ---------------------------------------------------------------------------
__global__ __launch_bounds__(256)
void slot_ln_qproj(const float* __restrict__ slots, const float* __restrict__ lw,
                   const float* __restrict__ lb, const float* __restrict__ Wq,
                   bf16_t* __restrict__ qbf) {
  __shared__ float sln[256];
  __shared__ float red[256];
  int r = blockIdx.x, t = threadIdx.x;
  int b = r >> 3, slot = r & 7;
  float v = slots[r * D_ + t];
  red[t] = v; __syncthreads();
  for (int s = 128; s > 0; s >>= 1) { if (t < s) red[t] += red[t + s]; __syncthreads(); }
  float mean = red[0] * (1.0f / D_); __syncthreads();
  red[t] = (v - mean) * (v - mean); __syncthreads();
  for (int s = 128; s > 0; s >>= 1) { if (t < s) red[t] += red[t + s]; __syncthreads(); }
  float inv = rsqrtf(red[0] * (1.0f / D_) + 1e-5f);
  sln[t] = (v - mean) * inv * lw[t] + lb[t];
  __syncthreads();
  float acc = 0.f;
  for (int i = 0; i < D_; ++i) acc += sln[i] * Wq[i * DI_ + t];
  int h = t >> 6, dh = t & 63;
  qbf[((b * HEADS_ + h) * 16 + slot) * DH_ + dh] = (bf16_t)acc;
}

// ---------------------------------------------------------------------------
// Attention: one (b,h) per block; 8 waves split 4096 tokens (512 each).
// dots = q@k^T via WMMA; softmax over slots is per-lane (8 acc VGPRs = 8 slots
// of one token); attn(+eps) staged in LDS -> updates += attn@v via WMMA.
// Normalization over tokens deferred (atomic rowsums + raw updates).
// ---------------------------------------------------------------------------
#define ATT_STRIDE 40
__global__ __launch_bounds__(256)
void slot_attention(const bf16_t* __restrict__ qbf, const bf16_t* __restrict__ Kbf,
                    const bf16_t* __restrict__ Vt, float* __restrict__ upd,
                    float* __restrict__ rowsum) {
  __shared__ bf16_t attn[8 * 16 * ATT_STRIDE];
  const int bh = blockIdx.x, b = bh >> 2, h = bh & 3;
  const int wave = threadIdx.x >> 5, lane = threadIdx.x & 31;
  const int lmod = lane & 15;
  const int koff = (lane < 16) ? 0 : 8;
  bf16_t* aw = &attn[wave * 16 * ATT_STRIDE];
  for (int i = lane; i < 16 * ATT_STRIDE; i += 32) aw[i] = (bf16_t)0.0f;

  const bf16_t* qr = qbf + (long)bh * 16 * DH_ + lmod * DH_;
  v16bf qa0 = frag16(*(const v8bf*)&qr[koff],      *(const v8bf*)&qr[koff + 16]);
  v16bf qa1 = frag16(*(const v8bf*)&qr[32 + koff], *(const v8bf*)&qr[32 + koff + 16]);

  v8f uacc[4] = {};
  float rs[8] = {};
  const float scale = 0.125f;  // DH^-0.5
  const long kbase = (long)b * N_ * DI_ + h * DH_;
  const long vbase = (long)(b * HEADS_ + h) * DH_ * N_;
  __syncthreads();

  for (int tc = wave * 512; tc < wave * 512 + 512; tc += 32) {
#pragma unroll
    for (int sub = 0; sub < 2; ++sub) {
      int t0 = tc + sub * 16;
      const bf16_t* kp = Kbf + kbase + (long)(t0 + lmod) * DI_;
      v16bf bk0 = frag16(*(const v8bf*)&kp[koff],      *(const v8bf*)&kp[koff + 16]);
      v16bf bk1 = frag16(*(const v8bf*)&kp[32 + koff], *(const v8bf*)&kp[32 + koff + 16]);
      v8f d = {};
      d = wmma_bf16(qa0, bk0, d);
      d = wmma_bf16(qa1, bk1, d);
      if (lane < 16) {  // lanes 0-15 hold slots 0..7 (valid) for token t0+lane
        float e[8];
        float mx = d[0] * scale;
#pragma unroll
        for (int i = 1; i < 8; ++i) mx = fmaxf(mx, d[i] * scale);
        float s = 0.f;
#pragma unroll
        for (int i = 0; i < 8; ++i) { e[i] = __expf(d[i] * scale - mx); s += e[i]; }
        float rcp = 1.0f / s;
#pragma unroll
        for (int i = 0; i < 8; ++i) {
          float a = e[i] * rcp + EPS_;
          rs[i] += a;
          aw[i * ATT_STRIDE + sub * 16 + lane] = (bf16_t)a;
        }
      }
    }
    __syncthreads();
    const bf16_t* ar = &aw[lmod * ATT_STRIDE];
    v16bf Af = frag16(*(const v8bf*)&ar[koff], *(const v8bf*)&ar[koff + 16]);
#pragma unroll
    for (int dt = 0; dt < 4; ++dt) {
      const bf16_t* vp = Vt + vbase + (long)(dt * 16 + lmod) * N_ + tc;
      v16bf Bf = frag16(*(const v8bf*)&vp[koff], *(const v8bf*)&vp[koff + 16]);
      uacc[dt] = wmma_bf16(Af, Bf, uacc[dt]);
    }
    __builtin_prefetch(Kbf + kbase + (long)(tc + 32 + lmod) * DI_, 0, 1);
    __syncthreads();
  }

#pragma unroll
  for (int i = 0; i < 8; ++i) {
    float tot = warp_sum(rs[i]);  // lanes 16-31 contribute zero
    if (lane == 0) atomicAdd(&rowsum[bh * SLOTS_ + i], tot);
  }
  if (lane < 16) {
#pragma unroll
    for (int dt = 0; dt < 4; ++dt)
#pragma unroll
      for (int vi = 0; vi < 8; ++vi)
        atomicAdd(&upd[((long)b * SLOTS_ + vi) * DI_ + h * DH_ + dt * 16 + lmod],
                  uacc[dt][vi]);
  }
}

// ---------------------------------------------------------------------------
// Normalize updates, @Wc, GRU cell, LN, MLP residual. One block per slot-row.
// ---------------------------------------------------------------------------
__global__ __launch_bounds__(256)
void slot_update(const float* __restrict__ upd_raw, const float* __restrict__ rowsum,
                 const float* __restrict__ slots_in, const float* __restrict__ Wc,
                 const float* __restrict__ w_ih, const float* __restrict__ w_hh,
                 const float* __restrict__ b_ih, const float* __restrict__ b_hh,
                 const float* __restrict__ lnm_w, const float* __restrict__ lnm_b,
                 const float* __restrict__ mlp_w1, const float* __restrict__ mlp_b1,
                 const float* __restrict__ mlp_w2, const float* __restrict__ mlp_b2,
                 float* __restrict__ slots_out, float* __restrict__ final_out) {
  __shared__ float u[256], c[256], hrow[256], mrow[256], hid[128], red[256];
  int r = blockIdx.x, t = threadIdx.x;
  int b = r >> 3, slot = r & 7;
  int h = t >> 6;
  float rsv = rowsum[(b * HEADS_ + h) * SLOTS_ + slot];
  u[t] = upd_raw[(long)r * DI_ + t] / rsv;
  hrow[t] = slots_in[(long)r * D_ + t];
  __syncthreads();
  float acc = 0.f;
  for (int i = 0; i < DI_; ++i) acc += u[i] * Wc[i * D_ + t];
  c[t] = acc;
  __syncthreads();
  float xr = b_ih[t], xz = b_ih[t + 256], xn = b_ih[t + 512];
  float hr = b_hh[t], hz = b_hh[t + 256], hn = b_hh[t + 512];
  for (int i = 0; i < D_; ++i) {
    float ci = c[i], hi = hrow[i];
    xr += ci * w_ih[t * D_ + i];
    xz += ci * w_ih[(t + 256) * D_ + i];
    xn += ci * w_ih[(t + 512) * D_ + i];
    hr += hi * w_hh[t * D_ + i];
    hz += hi * w_hh[(t + 256) * D_ + i];
    hn += hi * w_hh[(t + 512) * D_ + i];
  }
  float rg = 1.f / (1.f + __expf(-(xr + hr)));
  float zg = 1.f / (1.f + __expf(-(xz + hz)));
  float ng = tanhf(xn + rg * hn);
  float sv = (1.f - zg) * ng + zg * hrow[t];
  red[t] = sv; __syncthreads();
  for (int s = 128; s > 0; s >>= 1) { if (t < s) red[t] += red[t + s]; __syncthreads(); }
  float mean = red[0] * (1.0f / D_); __syncthreads();
  red[t] = (sv - mean) * (sv - mean); __syncthreads();
  for (int s = 128; s > 0; s >>= 1) { if (t < s) red[t] += red[t + s]; __syncthreads(); }
  float inv = rsqrtf(red[0] * (1.0f / D_) + 1e-5f);
  mrow[t] = (sv - mean) * inv * lnm_w[t] + lnm_b[t];
  __syncthreads();
  if (t < HID_) {
    float a = mlp_b1[t];
    for (int i = 0; i < D_; ++i) a += mrow[i] * mlp_w1[i * HID_ + t];
    hid[t] = fmaxf(a, 0.f);
  }
  __syncthreads();
  float o = mlp_b2[t];
  for (int k = 0; k < HID_; ++k) o += hid[k] * mlp_w2[k * D_ + t];
  o += sv;
  slots_out[(long)r * D_ + t] = o;
  if (final_out) final_out[(long)r * D_ + t] = o;
}

// ---------------------------------------------------------------------------
extern "C" void kernel_launch(void* const* d_in, const int* in_sizes, int n_in,
                              void* d_out, int out_size, void* d_ws, size_t ws_size,
                              hipStream_t stream) {
  const float* inputs   = (const float*)d_in[0];
  const float* mu       = (const float*)d_in[1];
  const float* logsigma = (const float*)d_in[2];
  const float* ln_in_w  = (const float*)d_in[3];
  const float* ln_in_b  = (const float*)d_in[4];
  const float* ln_s_w   = (const float*)d_in[5];
  const float* ln_s_b   = (const float*)d_in[6];
  const float* Wq       = (const float*)d_in[7];
  const float* Wk       = (const float*)d_in[8];
  const float* Wv       = (const float*)d_in[9];
  const float* Wc       = (const float*)d_in[10];
  const float* g_ih     = (const float*)d_in[11];
  const float* g_hh     = (const float*)d_in[12];
  const float* gb_ih    = (const float*)d_in[13];
  const float* gb_hh    = (const float*)d_in[14];
  const float* lnm_w    = (const float*)d_in[15];
  const float* lnm_b    = (const float*)d_in[16];
  const float* m_w1     = (const float*)d_in[17];
  const float* m_b1     = (const float*)d_in[18];
  const float* m_w2     = (const float*)d_in[19];
  const float* m_b2     = (const float*)d_in[20];

  char* ws = (char*)d_ws;
  size_t off = 0;
  bf16_t* Kbf = (bf16_t*)(ws + off); off += (size_t)B_ * N_ * DI_ * 2;          // 64 MB
  bf16_t* Vt  = (bf16_t*)(ws + off); off += (size_t)B_ * N_ * DI_ * 2;          // 64 MB
  bf16_t* WkT = (bf16_t*)(ws + off); off += (size_t)D_ * DI_ * 2;
  bf16_t* WvT = (bf16_t*)(ws + off); off += (size_t)D_ * DI_ * 2;
  float*  slots = (float*)(ws + off); off += (size_t)B_ * SLOTS_ * D_ * 4;
  bf16_t* qbf = (bf16_t*)(ws + off); off += (size_t)B_ * HEADS_ * 16 * DH_ * 2;
  float*  upd = (float*)(ws + off);  off += (size_t)B_ * SLOTS_ * DI_ * 4;
  float*  rowsum = (float*)(ws + off); off += (size_t)B_ * HEADS_ * SLOTS_ * 4;

  convert_weights<<<(D_ * DI_ + 255) / 256, 256, 0, stream>>>(Wk, Wv, WkT, WvT);
  init_slots<<<(B_ * SLOTS_ * D_ + 255) / 256, 256, 0, stream>>>(mu, logsigma, slots);
  ln_kv_proj<<<(B_ * N_) / 32, 256, 0, stream>>>(inputs, ln_in_w, ln_in_b, WkT, WvT, Kbf, Vt);

  for (int it = 0; it < ITERS_; ++it) {
    zero_ws<<<(B_ * SLOTS_ * DI_ + 255) / 256, 256, 0, stream>>>(upd, rowsum, (unsigned*)qbf);
    slot_ln_qproj<<<B_ * SLOTS_, 256, 0, stream>>>(slots, ln_s_w, ln_s_b, Wq, qbf);
    slot_attention<<<B_ * HEADS_, 256, 0, stream>>>(qbf, Kbf, Vt, upd, rowsum);
    slot_update<<<B_ * SLOTS_, 256, 0, stream>>>(
        upd, rowsum, slots, Wc, g_ih, g_hh, gb_ih, gb_hh, lnm_w, lnm_b,
        m_w1, m_b1, m_w2, m_b2, slots,
        (it == ITERS_ - 1) ? (float*)d_out : (float*)nullptr);
  }
}
